// cnnblstm_48086453846541
// MI455X (gfx1250) — matrix operand order repaired and checked by reading, not compiled, not run on hardware
//
#include <hip/hip_runtime.h>
#include <hip/hip_bf16.h>
#include <math.h>

typedef __attribute__((ext_vector_type(16))) _Float16 v16h;
typedef __attribute__((ext_vector_type(8)))  float    v8f;

#define BATCH 256
#define T_POOL 393
#define C_CH 128
#define HID 150
#define G4 600          // 4*HID
#define NP 608          // N padded to 38*16
#define KP 160          // recurrent K padded to 5*32
#define MROWS (BATCH * T_POOL)   // 100608, multiple of 16
#define MTILES (MROWS / 16)      // 6288
#define NTILES (NP / 16)         // 38
#define NPAIRS (NTILES / 2)      // 19

// ---- workspace layout (bytes, all 256B aligned) ----
#define OFF_XSEQ   ((size_t)0)                               // f16 [MROWS][128]   = 25,755,648
#define OFF_PRE    ((size_t)25755648)                        // f16 [2][MROWS][600]= 241,459,200
#define OFF_WHH    ((size_t)267214848)                       // f16 [2][608][160]  = 389,120
#define OFF_WIH    ((size_t)267603968)                       // f16 [2][608][128]  = 311,296
#define OFF_BSUM   ((size_t)267915264)                       // f32 [2][608]       = 4,864
#define OFF_POOLED ((size_t)267920128)                       // f32 [256][300]     = 307,200

__device__ __forceinline__ float hsig(float x) {
    return fminf(fmaxf(0.2f * x + 0.5f, 0.0f), 1.0f);
}

// ---------------------------------------------------------------------------
// Kernel 1: Conv1d(VALID,k=15) + bias + ReLU + MaxPool1d(2) + transpose -> f16
// grid (393, 256), block 128 (one thread per output channel)
// ---------------------------------------------------------------------------
__global__ __launch_bounds__(128) void k_conv(const float* __restrict__ x,
                                              const float* __restrict__ w,
                                              const float* __restrict__ bias,
                                              _Float16* __restrict__ xseq) {
    const int t = blockIdx.x;       // pooled time 0..392
    const int b = blockIdx.y;       // batch
    const int c = threadIdx.x;      // channel
    __shared__ float xs[3][16];
    if (threadIdx.x < 48) {
        const int f = threadIdx.x >> 4, i = threadIdx.x & 15;
        xs[f][i] = x[((size_t)b * 3 + f) * 800 + 2 * t + i];
    }
    __syncthreads();
    float s0 = 0.f, s1 = 0.f;
    const float* wc = w + c * 45;
    #pragma unroll
    for (int f = 0; f < 3; ++f)
        #pragma unroll
        for (int k = 0; k < 15; ++k) {
            const float wv = wc[f * 15 + k];
            s0 = fmaf(wv, xs[f][k], s0);
            s1 = fmaf(wv, xs[f][k + 1], s1);
        }
    const float bv = bias[c];
    s0 = fmaxf(s0 + bv, 0.f);
    s1 = fmaxf(s1 + bv, 0.f);
    xseq[((size_t)b * T_POOL + t) * C_CH + c] = (_Float16)fmaxf(s0, s1);
}

// ---------------------------------------------------------------------------
// Kernel 2: pack / pad weights to f16, precompute bias sums
// ---------------------------------------------------------------------------
__global__ __launch_bounds__(256) void k_prep(
        const float* __restrict__ wih_f, const float* __restrict__ whh_f,
        const float* __restrict__ bih_f, const float* __restrict__ bhh_f,
        const float* __restrict__ wih_b, const float* __restrict__ whh_b,
        const float* __restrict__ bih_b, const float* __restrict__ bhh_b,
        _Float16* __restrict__ wih16, _Float16* __restrict__ whh16,
        float* __restrict__ bsum) {
    const size_t i0 = (size_t)blockIdx.x * blockDim.x + threadIdx.x;
    const size_t stride = (size_t)gridDim.x * blockDim.x;
    for (size_t i = i0; i < (size_t)2 * NP * KP; i += stride) {
        const int dir = (int)(i / (NP * KP));
        const int rem = (int)(i % (NP * KP));
        const int n = rem / KP, k = rem % KP;
        const float* src = dir ? whh_b : whh_f;
        whh16[i] = (_Float16)((n < G4 && k < HID) ? src[(size_t)n * HID + k] : 0.f);
    }
    for (size_t i = i0; i < (size_t)2 * NP * C_CH; i += stride) {
        const int dir = (int)(i / (NP * C_CH));
        const int rem = (int)(i % (NP * C_CH));
        const int n = rem / C_CH, k = rem % C_CH;
        const float* src = dir ? wih_b : wih_f;
        wih16[i] = (_Float16)((n < G4) ? src[(size_t)n * C_CH + k] : 0.f);
    }
    for (size_t i = i0; i < (size_t)2 * NP; i += stride) {
        const int dir = (int)(i / NP), n = (int)(i % NP);
        bsum[i] = (n < G4) ? (dir ? (bih_b[n] + bhh_b[n]) : (bih_f[n] + bhh_f[n])) : 0.f;
    }
}

// ---------------------------------------------------------------------------
// Kernel 3: input projection GEMM (both directions):
//   pre[dir][m][n] = sum_k xseq[m][k] * w_ih[dir][n][k] + b_ih[n] + b_hh[n]
// M=100608, N=600(->608), K=128. Each wave computes a 16x32 output block
// (one m-tile x two adjacent n-tiles) so the A fragment is reused across
// two WMMA chains: 8x v_wmma_f32_16x16x32_f16 per wave job.
// ---------------------------------------------------------------------------
__global__ __launch_bounds__(256) void k_inproj(const _Float16* __restrict__ xseq,
                                                const _Float16* __restrict__ wih,
                                                const float* __restrict__ bsum,
                                                _Float16* __restrict__ pre) {
    const int lane = threadIdx.x & 31;
    const int wave = threadIdx.x >> 5;
    const int dir = blockIdx.y;
    const long job = (long)blockIdx.x * 8 + wave;
    const long total = (long)MTILES * NPAIRS;       // 119,472
    if (job >= total) return;                       // wave-uniform exit
    const int mtile = (int)(job / NPAIRS);
    const int ntile0 = (int)(job % NPAIRS) * 2;

    // WMMA f16 lane mappings (ISA 7.12.2)
    const int rowA  = lane & 15;
    const int koffA = (lane < 16) ? 0 : 16;         // A: K0-15 vs K16-31 (contiguous)
    const int colB  = lane & 15;
    const int koffB = (lane < 16) ? 0 : 16;         // B: K0-15 vs K16-31 (contiguous)
    const int n0 = ntile0 * 16 + colB;              // <= 591, always valid
    const int n1 = n0 + 16;                         // may reach 607, guard on store

    v8f acc0, acc1;
    const float bv0 = bsum[dir * NP + n0];
    const float bv1 = bsum[dir * NP + n1];
    #pragma unroll
    for (int i = 0; i < 8; ++i) { acc0[i] = bv0; acc1[i] = bv1; }

    const _Float16* arow  = xseq + (size_t)(mtile * 16 + rowA) * C_CH;
    const _Float16* brow0 = wih + (size_t)dir * NP * C_CH + (size_t)n0 * C_CH;
    const _Float16* brow1 = brow0 + (size_t)16 * C_CH;
    #pragma unroll
    for (int kt = 0; kt < 4; ++kt) {
        v16h a, b0, b1;
        const int ka = kt * 32 + koffA;
        const int kb = kt * 32 + koffB;
        ((uint4*)&a)[0]  = *(const uint4*)(arow + ka);
        ((uint4*)&a)[1]  = *(const uint4*)(arow + ka + 8);
        ((uint4*)&b0)[0] = *(const uint4*)(brow0 + kb);
        ((uint4*)&b0)[1] = *(const uint4*)(brow0 + kb + 8);
        ((uint4*)&b1)[0] = *(const uint4*)(brow1 + kb);
        ((uint4*)&b1)[1] = *(const uint4*)(brow1 + kb + 8);
        acc0 = __builtin_amdgcn_wmma_f32_16x16x32_f16(false, a, false, b0,
                                                      (short)0, acc0, false, false);
        acc1 = __builtin_amdgcn_wmma_f32_16x16x32_f16(false, a, false, b1,
                                                      (short)0, acc1, false, false);
    }
    const int mbase = mtile * 16 + ((lane < 16) ? 0 : 8);
    _Float16* pd = pre + (size_t)dir * MROWS * G4;
    #pragma unroll
    for (int r = 0; r < 8; ++r)
        pd[(size_t)(mbase + r) * G4 + n0] = (_Float16)acc0[r];
    if (n1 < G4) {
        #pragma unroll
        for (int r = 0; r < 8; ++r)
            pd[(size_t)(mbase + r) * G4 + n1] = (_Float16)acc1[r];
    }
}

// ---------------------------------------------------------------------------
// Kernel 4: persistent BiLSTM recurrence + fused time-max-pool.
// grid = 32 (16 batch slices x 2 dirs), block = 256 (8 waves).
// Per step: D[16,608] = h[16,160] @ Whh^T (WMMA, K padded), then gate math,
// update c/h/runmax in LDS. Two barriers per step. Next step's preactivation
// slice is prefetched (global_prefetch_b8) during the gate phase to hide
// HBM latency behind the barrier + WMMA work.
// ---------------------------------------------------------------------------
__global__ __launch_bounds__(256) void k_lstm(const _Float16* __restrict__ pre,
                                              const _Float16* __restrict__ whh,
                                              float* __restrict__ pooled) {
    __shared__ _Float16 h_lds[16 * KP];     //  5,120 B
    __shared__ float    D_lds[16 * NP];     // 38,912 B
    __shared__ float    c_lds[16 * 152];    //  9,728 B
    __shared__ float    m_lds[16 * 152];    //  9,728 B  (total 63,488 B)

    const int tid = threadIdx.x;
    const int lane = tid & 31, wave = tid >> 5;
    const int dir = blockIdx.x & 1;
    const int b0 = (blockIdx.x >> 1) * 16;

    for (int i = tid; i < 16 * KP; i += 256) h_lds[i] = (_Float16)0.f;
    for (int i = tid; i < 16 * 152; i += 256) { c_lds[i] = 0.f; m_lds[i] = -1e30f; }
    __syncthreads();

    const _Float16* Wd = whh + (size_t)dir * NP * KP;
    const _Float16* pd = pre + (size_t)dir * MROWS * G4;

    const int rowA  = lane & 15;
    const int koffA = (lane < 16) ? 0 : 16;
    const int colB  = lane & 15;
    const int koffB = (lane < 16) ? 0 : 16;
    const int mloc  = (lane < 16) ? 0 : 8;

    // Prefetch step 0's preactivation slice (16 rows x 1200 B).
    {
        const int t0 = dir ? (T_POOL - 1) : 0;
        for (int q = tid; q < 160; q += 256) {
            const int b = q / 10, seg = q - b * 10;
            if (seg * 64 < G4)
                __builtin_prefetch(pd + ((size_t)(b0 + b) * T_POOL + t0) * G4 + seg * 64, 0, 3);
        }
    }

    for (int ts = 0; ts < T_POOL; ++ts) {
        const int t = dir ? (T_POOL - 1 - ts) : ts;

        // ---- recurrent GEMM: D = h @ Whh^T ----
        for (int nt = wave; nt < NTILES; nt += 8) {
            v8f acc;
            #pragma unroll
            for (int i = 0; i < 8; ++i) acc[i] = 0.f;
            const _Float16* brow = Wd + (size_t)(nt * 16 + colB) * KP;
            #pragma unroll
            for (int kt = 0; kt < 5; ++kt) {
                v16h a, b;
                const int ka = kt * 32 + koffA;
                const int kb = kt * 32 + koffB;
                ((uint4*)&a)[0] = *(const uint4*)(h_lds + rowA * KP + ka);
                ((uint4*)&a)[1] = *(const uint4*)(h_lds + rowA * KP + ka + 8);
                ((uint4*)&b)[0] = *(const uint4*)(brow + kb);
                ((uint4*)&b)[1] = *(const uint4*)(brow + kb + 8);
                acc = __builtin_amdgcn_wmma_f32_16x16x32_f16(false, a, false, b,
                                                             (short)0, acc, false, false);
            }
            #pragma unroll
            for (int r = 0; r < 8; ++r)
                D_lds[(mloc + r) * NP + nt * 16 + colB] = acc[r];
        }
        __syncthreads();

        // ---- prefetch next step's preactivation slice ----
        if (ts + 1 < T_POOL) {
            const int tn = dir ? (T_POOL - 2 - ts) : (ts + 1);
            for (int q = tid; q < 160; q += 256) {
                const int b = q / 10, seg = q - b * 10;
                if (seg * 64 < G4)
                    __builtin_prefetch(pd + ((size_t)(b0 + b) * T_POOL + tn) * G4 + seg * 64, 0, 3);
            }
        }

        // ---- gate nonlinearity + state update + running max ----
        for (int j = tid; j < 16 * HID; j += 256) {
            const int b = j / HID, hi = j - b * HID;
            const size_t prow = ((size_t)(b0 + b) * T_POOL + t) * (size_t)G4;
            const float gi = D_lds[b * NP + hi]       + (float)pd[prow + hi];
            const float gf = D_lds[b * NP + 150 + hi] + (float)pd[prow + 150 + hi];
            const float gg = D_lds[b * NP + 300 + hi] + (float)pd[prow + 300 + hi];
            const float go = D_lds[b * NP + 450 + hi] + (float)pd[prow + 450 + hi];
            const float c = hsig(gf) * c_lds[b * 152 + hi] + hsig(gi) * tanhf(gg);
            const float h = hsig(go) * tanhf(c);
            c_lds[b * 152 + hi] = c;
            if (h > m_lds[b * 152 + hi]) m_lds[b * 152 + hi] = h;
            h_lds[b * KP + hi] = (_Float16)h;
        }
        __syncthreads();
    }

    for (int j = tid; j < 16 * HID; j += 256) {
        const int b = j / HID, hi = j - b * HID;
        pooled[(size_t)(b0 + b) * 300 + dir * HID + hi] = m_lds[b * 152 + hi];
    }
}

// ---------------------------------------------------------------------------
// Kernel 5: MLP head + softmax. One block per batch row.
// ---------------------------------------------------------------------------
__global__ __launch_bounds__(64) void k_head(const float* __restrict__ pooled,
                                             const float* __restrict__ w1,
                                             const float* __restrict__ b1,
                                             const float* __restrict__ w2,
                                             const float* __restrict__ b2,
                                             float* __restrict__ out) {
    const int b = blockIdx.x;
    const int tid = threadIdx.x;
    __shared__ float z[50];
    __shared__ float lg[10];
    const float* p = pooled + (size_t)b * 300;
    if (tid < 50) {
        float s = b1[tid];
        const float* wr = w1 + tid * 300;
        for (int k = 0; k < 300; ++k) s = fmaf(wr[k], p[k], s);
        z[tid] = fmaxf(s, 0.f);
    }
    __syncthreads();
    if (tid < 10) {
        float s = b2[tid];
        const float* wr = w2 + tid * 50;
        for (int k = 0; k < 50; ++k) s = fmaf(wr[k], z[k], s);
        lg[tid] = s;
    }
    __syncthreads();
    if (tid == 0) {
        float mx = lg[0];
        for (int k = 1; k < 10; ++k) mx = fmaxf(mx, lg[k]);
        float e[10], sum = 0.f;
        for (int k = 0; k < 10; ++k) { e[k] = expf(lg[k] - mx); sum += e[k]; }
        const float inv = 1.0f / sum;
        for (int k = 0; k < 10; ++k) out[(size_t)b * 10 + k] = e[k] * inv;
    }
}

// ---------------------------------------------------------------------------
extern "C" void kernel_launch(void* const* d_in, const int* in_sizes, int n_in,
                              void* d_out, int out_size, void* d_ws, size_t ws_size,
                              hipStream_t stream) {
    const float* x      = (const float*)d_in[0];
    const float* conv_w = (const float*)d_in[1];
    const float* conv_b = (const float*)d_in[2];
    const float* w_ih_f = (const float*)d_in[3];
    const float* w_hh_f = (const float*)d_in[4];
    const float* b_ih_f = (const float*)d_in[5];
    const float* b_hh_f = (const float*)d_in[6];
    const float* w_ih_b = (const float*)d_in[7];
    const float* w_hh_b = (const float*)d_in[8];
    const float* b_ih_b = (const float*)d_in[9];
    const float* b_hh_b = (const float*)d_in[10];
    const float* lin1_w = (const float*)d_in[11];
    const float* lin1_b = (const float*)d_in[12];
    const float* lin2_w = (const float*)d_in[13];
    const float* lin2_b = (const float*)d_in[14];

    char* ws = (char*)d_ws;
    _Float16* xseq16 = (_Float16*)(ws + OFF_XSEQ);
    _Float16* pre16  = (_Float16*)(ws + OFF_PRE);
    _Float16* whh16  = (_Float16*)(ws + OFF_WHH);
    _Float16* wih16  = (_Float16*)(ws + OFF_WIH);
    float*    bsum   = (float*)(ws + OFF_BSUM);
    float*    pooledp= (float*)(ws + OFF_POOLED);

    k_conv<<<dim3(T_POOL, BATCH), 128, 0, stream>>>(x, conv_w, conv_b, xseq16);
    k_prep<<<512, 256, 0, stream>>>(w_ih_f, w_hh_f, b_ih_f, b_hh_f,
                                    w_ih_b, w_hh_b, b_ih_b, b_hh_b,
                                    wih16, whh16, bsum);
    const long total_jobs = (long)MTILES * NPAIRS;            // 119,472
    dim3 g2((unsigned)((total_jobs + 7) / 8), 2);
    k_inproj<<<g2, 256, 0, stream>>>(xseq16, wih16, bsum, pre16);
    k_lstm<<<32, 256, 0, stream>>>(pre16, whh16, pooledp);
    k_head<<<BATCH, 64, 0, stream>>>(pooledp, lin1_w, lin1_b, lin2_w, lin2_b,
                                     (float*)d_out);
}
